// Retention_32727650795919
// MI455X (gfx1250) — matrix operand order repaired and checked by reading, not compiled, not run on hardware
//
#include <hip/hip_runtime.h>
#include <hip/hip_bf16.h>

// ---------------------------------------------------------------------------
// Retention block for MI455X (gfx1250, wave32, WMMA + TDM).
//   B=4, S=2048, D=1024, H=16, HD=64
// Pipeline:
//   K1: q',k' (rotary-expanded to 128 feat, bf16) and v (bf16) via WMMA GEMM
//   K2: flash-style causal decayed retention + LayerNorm + SiLU
//       (WMMA x2, TDM double-buffered K/V staging)
//   K3: output projection (WMMA, TDM-staged A operand), fp32 out
// ---------------------------------------------------------------------------

#define B_  4
#define S_  2048
#define D_  1024
#define H_  16
#define HD_ 64

typedef __attribute__((ext_vector_type(16))) __bf16 v16bf;
typedef __attribute__((ext_vector_type(8)))  float  v8f;
typedef unsigned int uint32x4 __attribute__((ext_vector_type(4)));
typedef int          int32x8  __attribute__((ext_vector_type(8)));
typedef int          int32x4  __attribute__((ext_vector_type(4)));

static __device__ inline v8f zero8() {
  v8f z = {0.f, 0.f, 0.f, 0.f, 0.f, 0.f, 0.f, 0.f};
  return z;
}

// WMMA wrapper: D = A(16x32 bf16) * B(32x16 bf16) + C(16x16 f32)
static __device__ inline v8f wmma_bf16(v16bf a, v16bf b, v8f c) {
  return __builtin_amdgcn_wmma_f32_16x16x32_bf16(
      /*neg_a=*/false, a, /*neg_b=*/false, b,
      /*c_mod=*/(short)0, c, /*reuse_a=*/false, /*reuse_b=*/false);
}

// A-fragment (16x32 bf16, MxK) from row-major [m][k] storage, stride ld.
static __device__ inline v16bf load_a_frag(const __bf16* src, int ld, int lane) {
  const int row  = lane & 15;
  const int half = lane >> 4;
  v16bf a;
#pragma unroll
  for (int p = 0; p < 8; ++p) {
    const int k0 = 16 * half + 2 * p;
    a[2 * p]     = src[row * ld + k0];
    a[2 * p + 1] = src[row * ld + k0 + 1];
  }
  return a;
}

// B-fragment (32x16 bf16, KxN) from [n][k] row-major storage (B^T, contiguous k).
static __device__ inline v16bf load_b_frag_nk(const __bf16* src, int ld, int lane) {
  const int n    = lane & 15;
  const int half = lane >> 4;
  v16bf b;
#pragma unroll
  for (int v = 0; v < 8; ++v) {
    const int k0 = 16 * half + 2 * v;
    b[2 * v]     = src[n * ld + k0];
    b[2 * v + 1] = src[n * ld + k0 + 1];
  }
  return b;
}

// B-fragment (32x16 bf16, KxN) from [k][n] row-major storage.
static __device__ inline v16bf load_b_frag_kn(const __bf16* src, int ld, int lane) {
  const int n    = lane & 15;
  const int half = lane >> 4;
  v16bf b;
#pragma unroll
  for (int v = 0; v < 8; ++v) {
    const int k0 = 16 * half + 2 * v;
    b[2 * v]     = src[k0 * ld + n];
    b[2 * v + 1] = src[(k0 + 1) * ld + n];
  }
  return b;
}

// Generic LDS byte offset of a __shared__ object (low 32 bits of generic addr).
static __device__ inline uint32_t lds_off(const void* p) {
  return (uint32_t)(uintptr_t)p;
}

// ---------------------------------------------------------------------------
// Tensor Data Mover: 2D tile load (8-byte elements) Global -> LDS.
// D# per CDNA5 ISA 8.3/8.4: group0 = {count|flags, lds_addr, global_addr, type},
// group1 = dims/strides. tile_w/pitch in 8B units; 1D copy: rows=1, pitch=w.
// This toolchain's builtin is the 6-arg form: (g0, g1, g2, g3, g4, cpol);
// only groups 0/1 are meaningful for a <=2D tile, the rest are zero.
// ---------------------------------------------------------------------------
static __device__ inline void tdm_load_2d(uint32_t lds_addr, const void* gptr,
                                          uint32_t tile_w_units,
                                          uint32_t tile_rows,
                                          uint32_t pitch_units) {
  const unsigned long long ga = (unsigned long long)(uintptr_t)gptr;
  uint32x4 g0;
  g0[0] = 1u;                                        // count=1, user descriptor
  g0[1] = lds_addr;                                  // LDS byte address
  g0[2] = (unsigned)(ga & 0xFFFFFFFFull);            // global_addr[31:0]
  g0[3] = (unsigned)((ga >> 32) & 0x01FFFFFFull)     // global_addr[56:32]
          | (2u << 30);                              // type=2 (image)
  int32x8 g1;
  g1[0] = (int)(3u << 16);                           // data_size=3 (8B), mask=0
  g1[1] = (int)((pitch_units & 0xFFFFu) << 16);      // tensor_dim0[15:0]
  g1[2] = (int)(((pitch_units >> 16) & 0xFFFFu)      // tensor_dim0[31:16]
                | (0xFFFFu << 16));                  // tensor_dim1[15:0] (big)
  g1[3] = (int)(0x7FFFu                              // tensor_dim1[31:16]
                | ((tile_w_units & 0xFFFFu) << 16)); // tile_dim0
  g1[4] = (int)(tile_rows & 0xFFFFu);                // tile_dim1 (tile_dim2=0)
  g1[5] = (int)pitch_units;                          // tensor_dim0_stride[31:0]
  g1[6] = 0;                                         // stride hi / dim1_stride lo
  g1[7] = 0;
  int32x4 zz = {0, 0, 0, 0};
  int32x8 z8 = {0, 0, 0, 0, 0, 0, 0, 0};
  __builtin_amdgcn_tensor_load_to_lds(g0, g1, zz, zz, z8, 0);
}

// ---------------------------------------------------------------------------
// Kernel 1: fused QKV projection + rotary expansion.
//   grid (32, 16, 3) x 256 threads. z=0:q, 1:k, 2:v.
//   Tile: M=256 (8 waves x 32 rows), N=64, K-step 32.
// ---------------------------------------------------------------------------
__global__ __launch_bounds__(256)
void qkv_rope_kernel(const float* __restrict__ x,
                     const float* __restrict__ Wq,
                     const float* __restrict__ Wk,
                     const float* __restrict__ Wv,
                     __bf16* __restrict__ qr,
                     __bf16* __restrict__ kr,
                     __bf16* __restrict__ vw) {
  __shared__ __align__(16) __bf16 As[256][32];
  __shared__ __align__(16) __bf16 Bs[64][32];

  const int m0 = blockIdx.x * 256;
  const int n0 = blockIdx.y * 64;
  const int z  = blockIdx.z;
  const float* __restrict__ W = (z == 0) ? Wq : (z == 1) ? Wk : Wv;

  const int tid  = threadIdx.x;
  const int lane = tid & 31;
  const int wave = tid >> 5;

  v8f acc[2][4];
#pragma unroll
  for (int i = 0; i < 2; ++i)
#pragma unroll
    for (int j = 0; j < 4; ++j) acc[i][j] = zero8();

  const int brow = tid >> 2;       // 0..63
  const int bseg = (tid & 3) * 8;  // 0,8,16,24

  for (int kk = 0; kk < D_; kk += 32) {
    // stage A: one row per thread, f32 -> bf16
    const float* xsrc = &x[(size_t)(m0 + tid) * D_ + kk];
    if (kk + 32 < D_) __builtin_prefetch(xsrc + 32, 0, 1);
#pragma unroll
    for (int j = 0; j < 32; ++j) As[tid][j] = (__bf16)xsrc[j];
    // stage B: weight tile, f32 -> bf16
    const float* wsrc = &W[(size_t)(n0 + brow) * D_ + kk + bseg];
    if (kk + 32 < D_) __builtin_prefetch(wsrc + 32, 0, 1);
#pragma unroll
    for (int j = 0; j < 8; ++j) Bs[brow][bseg + j] = (__bf16)wsrc[j];
    __syncthreads();

    const v16bf a0 = load_a_frag(&As[wave * 32][0], 32, lane);
    const v16bf a1 = load_a_frag(&As[wave * 32 + 16][0], 32, lane);
#pragma unroll
    for (int nt = 0; nt < 4; ++nt) {
      const v16bf b = load_b_frag_nk(&Bs[nt * 16][0], 32, lane);
      acc[0][nt] = wmma_bf16(a0, b, acc[0][nt]);
      acc[1][nt] = wmma_bf16(a1, b, acc[1][nt]);
    }
    __syncthreads();
  }

  // epilogue: scatter with rotary weighting
  const float NEG_L2_1E4_OVER_32 = -0.41524101298491f;  // -log2(10000)/32
  const int col = lane & 15;
  const int rbase = (lane >> 4) * 8;
#pragma unroll
  for (int mr = 0; mr < 2; ++mr) {
#pragma unroll
    for (int nt = 0; nt < 4; ++nt) {
#pragma unroll
      for (int r = 0; r < 8; ++r) {
        const int m = m0 + wave * 32 + mr * 16 + rbase + r;  // 0..8191
        const int n = n0 + nt * 16 + col;                    // 0..1023
        const int b = m >> 11;
        const int s = m & (S_ - 1);
        const int h = n >> 6;
        const int d = n & 63;
        const float val = acc[mr][nt][r];
        if (z == 2) {
          vw[((size_t)(b * H_ + h) * S_ + s) * HD_ + d] = (__bf16)val;
        } else {
          const float theta = exp2f((float)d * NEG_L2_1E4_OVER_32);
          const float ang   = (float)s * theta;
          __bf16* dst = (z == 0) ? qr : kr;
          const size_t base = ((size_t)(b * H_ + h) * S_ + s) * 128;
          dst[base + d]      = (__bf16)(val * __cosf(ang));
          dst[base + 64 + d] = (__bf16)(val * __sinf(ang));
        }
      }
    }
  }
}

// ---------------------------------------------------------------------------
// Kernel 2: causal decayed retention + LayerNorm + SiLU (flash-style).
//   grid (32, 64) x 128 threads (4 waves). x = 64-query tile, y = b*H+h.
//   K'/V tiles double-buffered via TDM; one barrier per key-step.
// ---------------------------------------------------------------------------
__global__ __launch_bounds__(128)
void retention_attn_kernel(const __bf16* __restrict__ qr,
                           const __bf16* __restrict__ kr,
                           const __bf16* __restrict__ vw,
                           const float* __restrict__ ln_w,
                           const float* __restrict__ ln_b,
                           __bf16* __restrict__ attn) {
  __shared__ __align__(16) __bf16 Qs[64][128];      // 16 KB
  __shared__ __align__(16) __bf16 Ks[2][32][128];   // 2 x 8 KB
  __shared__ __align__(16) __bf16 Vs[2][32][64];    // 2 x 4 KB
  __shared__ __align__(16) __bf16 Ps[4][16][32];    // 4 KB

  const int qt = blockIdx.x;   // 0..31
  const int bh = blockIdx.y;   // 0..63
  const int h  = bh & 15;
  const int tid = threadIdx.x, lane = tid & 31, wave = tid >> 5;
  const int q0 = qt * 64;

  const size_t qk_base = (size_t)bh * S_ * 128;
  const size_t v_base  = (size_t)bh * S_ * HD_;

  const float gamma = 1.0f - exp2f(-5.0f - (float)h);
  const float log2g = log2f(gamma);

  const int nsteps = 2 * qt + 2;

  // prologue: TDM loads of Q tile (16 KB) and step-0 K'/V tiles
  if (wave == 0) {
    tdm_load_2d(lds_off(&Qs[0][0]),    qr + qk_base + (size_t)q0 * 128, 2048, 1, 2048);
    tdm_load_2d(lds_off(&Ks[0][0][0]), kr + qk_base,                   1024, 1, 1024);
    tdm_load_2d(lds_off(&Vs[0][0][0]), vw + v_base,                     512, 1,  512);
    __builtin_amdgcn_s_wait_tensorcnt(0);
  }
  __syncthreads();

  // preload Q A-fragments (K=128 -> 4 frags of 16x32)
  v16bf qa[4];
#pragma unroll
  for (int kk = 0; kk < 4; ++kk)
    qa[kk] = load_a_frag(&Qs[wave * 16][kk * 32], 128, lane);

  v8f oacc[4];
#pragma unroll
  for (int i = 0; i < 4; ++i) oacc[i] = zero8();

  const int qrow_hi = q0 + wave * 16 + 15;
  const int rbase = (lane >> 4) * 8;
  const int col   = lane & 15;

  for (int step = 0; step < nsteps; ++step) {
    const int t = step * 32;
    const int cur = step & 1;

    // prefetch next K'/V tiles into the other buffer while computing this one
    if (wave == 0 && step + 1 < nsteps) {
      const int tn = t + 32;
      tdm_load_2d(lds_off(&Ks[cur ^ 1][0][0]),
                  kr + qk_base + (size_t)tn * 128, 1024, 1, 1024);
      tdm_load_2d(lds_off(&Vs[cur ^ 1][0][0]),
                  vw + v_base + (size_t)tn * HD_, 512, 1, 512);
    }

    if (t <= qrow_hi) {  // this wave still has causal work in this key tile
      // scores: two 16x16 tiles over K=128
      v8f sc0 = zero8(), sc1 = zero8();
#pragma unroll
      for (int kk = 0; kk < 4; ++kk) {
        const v16bf b0 = load_b_frag_nk(&Ks[cur][0][kk * 32], 128, lane);
        const v16bf b1 = load_b_frag_nk(&Ks[cur][16][kk * 32], 128, lane);
        sc0 = wmma_bf16(qa[kk], b0, sc0);
        sc1 = wmma_bf16(qa[kk], b1, sc1);
      }
      // causal exponential decay; stage P (16x32 bf16) through per-wave LDS
#pragma unroll
      for (int r = 0; r < 8; ++r) {
        const int i = q0 + wave * 16 + rbase + r;
        const int j0 = t + col;
        const int j1 = t + 16 + col;
        const float w0 = (j0 <= i) ? exp2f((float)(i - j0) * log2g) : 0.0f;
        const float w1 = (j1 <= i) ? exp2f((float)(i - j1) * log2g) : 0.0f;
        Ps[wave][rbase + r][col]      = (__bf16)(sc0[r] * w0);
        Ps[wave][rbase + r][16 + col] = (__bf16)(sc1[r] * w1);
      }
      asm volatile("s_wait_dscnt 0" ::: "memory");  // LDS RAW within wave
      // O += P (16x32) x V (32x64)
      const v16bf pa = load_a_frag(&Ps[wave][0][0], 32, lane);
#pragma unroll
      for (int n4 = 0; n4 < 4; ++n4) {
        const v16bf bv = load_b_frag_kn(&Vs[cur][0][n4 * 16], HD_, lane);
        oacc[n4] = wmma_bf16(pa, bv, oacc[n4]);
      }
    }

    // wave0 waits for the next buffer's DMA; barrier releases everyone and
    // also makes this iteration's reads complete before the buffer is reused.
    if (wave == 0) __builtin_amdgcn_s_wait_tensorcnt(0);
    __syncthreads();
  }

  // epilogue: LayerNorm over HD=64 (half-wave reductions) + SiLU, bf16 store
  const int b = bh >> 4;
#pragma unroll
  for (int r = 0; r < 8; ++r) {
    float vals[4], sum = 0.f, sq = 0.f;
#pragma unroll
    for (int n4 = 0; n4 < 4; ++n4) {
      vals[n4] = oacc[n4][r];
      sum += vals[n4];
      sq  += vals[n4] * vals[n4];
    }
#pragma unroll
    for (int m = 1; m < 16; m <<= 1) {
      sum += __shfl_xor(sum, m, 32);
      sq  += __shfl_xor(sq, m, 32);
    }
    const float mu   = sum * (1.0f / 64.0f);
    const float var  = sq * (1.0f / 64.0f) - mu * mu;
    const float rstd = rsqrtf(var + 1e-5f);
    const int s = q0 + wave * 16 + rbase + r;
    const size_t orow = ((size_t)b * S_ + s) * D_ + h * HD_;
#pragma unroll
    for (int n4 = 0; n4 < 4; ++n4) {
      const int d = n4 * 16 + col;
      float y = (vals[n4] - mu) * rstd * ln_w[d] + ln_b[d];
      y = y / (1.0f + __expf(-y));  // SiLU
      attn[orow + d] = (__bf16)y;
    }
  }
}

// ---------------------------------------------------------------------------
// Kernel 3: output projection out = attn(bf16) @ Wo.T, fp32 result.
//   grid (32, 16) x 256 threads, M=256 tile; A operand staged by TDM (2D tile:
//   256 rows x 64 B, pitch 2048 B), B converted f32->bf16 by the VALU path.
// ---------------------------------------------------------------------------
__global__ __launch_bounds__(256)
void out_proj_kernel(const __bf16* __restrict__ a,
                     const float* __restrict__ Wo,
                     float* __restrict__ out) {
  __shared__ __align__(16) __bf16 As[256][32];
  __shared__ __align__(16) __bf16 Bs[64][32];

  const int m0 = blockIdx.x * 256;
  const int n0 = blockIdx.y * 64;
  const int tid = threadIdx.x, lane = tid & 31, wave = tid >> 5;

  v8f acc[2][4];
#pragma unroll
  for (int i = 0; i < 2; ++i)
#pragma unroll
    for (int j = 0; j < 4; ++j) acc[i][j] = zero8();

  const int brow = tid >> 2;
  const int bseg = (tid & 3) * 8;

  for (int kk = 0; kk < D_; kk += 32) {
    // stage A via TDM: 256 rows x 8 units(64B), pitch 256 units(2048B)
    if (wave == 0)
      tdm_load_2d(lds_off(&As[0][0]), a + (size_t)m0 * D_ + kk, 8, 256, 256);
    // stage B: Wo f32 -> bf16
    const float* wsrc = &Wo[(size_t)(n0 + brow) * D_ + kk + bseg];
    if (kk + 32 < D_) __builtin_prefetch(wsrc + 32, 0, 1);
#pragma unroll
    for (int j = 0; j < 8; ++j) Bs[brow][bseg + j] = (__bf16)wsrc[j];
    if (wave == 0) __builtin_amdgcn_s_wait_tensorcnt(0);
    __syncthreads();

    const v16bf a0 = load_a_frag(&As[wave * 32][0], 32, lane);
    const v16bf a1 = load_a_frag(&As[wave * 32 + 16][0], 32, lane);
#pragma unroll
    for (int nt = 0; nt < 4; ++nt) {
      const v16bf bf = load_b_frag_nk(&Bs[nt * 16][0], 32, lane);
      acc[0][nt] = wmma_bf16(a0, bf, acc[0][nt]);
      acc[1][nt] = wmma_bf16(a1, bf, acc[1][nt]);
    }
    __syncthreads();
  }

  const int col = lane & 15;
  const int rbase = (lane >> 4) * 8;
#pragma unroll
  for (int mr = 0; mr < 2; ++mr) {
#pragma unroll
    for (int nt = 0; nt < 4; ++nt) {
#pragma unroll
      for (int r = 0; r < 8; ++r) {
        const int m = m0 + wave * 32 + mr * 16 + rbase + r;
        const int n = n0 + nt * 16 + col;
        out[(size_t)m * D_ + n] = acc[mr][nt][r];
      }
    }
  }
}

// ---------------------------------------------------------------------------
// Launch. Workspace layout (bytes):
//   [0, 32M)   q' bf16 [B,H,S,128]
//   [32M,64M)  k' bf16 [B,H,S,128]
//   [64M,80M)  v  bf16 [B,H,S,64]
//   [80M,96M)  attn bf16 [B,S,D]
// ---------------------------------------------------------------------------
extern "C" void kernel_launch(void* const* d_in, const int* in_sizes, int n_in,
                              void* d_out, int out_size, void* d_ws,
                              size_t ws_size, hipStream_t stream) {
  (void)in_sizes; (void)n_in; (void)out_size; (void)ws_size;
  const float* x    = (const float*)d_in[0];
  const float* Wq   = (const float*)d_in[1];
  const float* Wk   = (const float*)d_in[2];
  const float* Wv   = (const float*)d_in[3];
  const float* Wo   = (const float*)d_in[4];
  const float* ln_w = (const float*)d_in[5];
  const float* ln_b = (const float*)d_in[6];
  float* out = (float*)d_out;

  char* ws = (char*)d_ws;
  __bf16* qr   = (__bf16*)(ws);
  __bf16* kr   = (__bf16*)(ws + ((size_t)32 << 20));
  __bf16* vw   = (__bf16*)(ws + ((size_t)64 << 20));
  __bf16* attn = (__bf16*)(ws + ((size_t)80 << 20));

  dim3 g1(32, 16, 3);
  qkv_rope_kernel<<<g1, 256, 0, stream>>>(x, Wq, Wk, Wv, qr, kr, vw);

  dim3 g2(32, 64);
  retention_attn_kernel<<<g2, 128, 0, stream>>>(qr, kr, vw, ln_w, ln_b, attn);

  dim3 g3(32, 16);
  out_proj_kernel<<<g3, 256, 0, stream>>>(attn, Wo, out);
}
